// MultiAttnMatch_38276748542485
// MI455X (gfx1250) — matrix operand order repaired and checked by reading, not compiled, not run on hardware
//
#include <hip/hip_runtime.h>
#include <cstdint>
#include <cstddef>

// ---------------------------------------------------------------------------
// Problem constants (fixed by the reference)
// ---------------------------------------------------------------------------
namespace cfg {
constexpr int B    = 8;
constexpr int L    = 1024;   // L1 == L2
constexpr int DIN  = 768;
constexpr int H    = 12;
constexpr int DH   = 64;     // DK == DV
constexpr int NP   = H * DH; // 768 projected width
}

typedef _Float16 v16h __attribute__((ext_vector_type(16)));
typedef _Float16 v8h  __attribute__((ext_vector_type(8)));
typedef float    v8f  __attribute__((ext_vector_type(8)));
typedef int      v4i  __attribute__((ext_vector_type(4)));

// ---------------------------------------------------------------------------
// CDNA5 async global->LDS copy (GLOBAL_LOAD_ASYNC_TO_LDS_B128, ASYNCcnt),
// with a synchronous fallback if the builtin is unavailable.
// Builtin signature (from the gfx1250 diagnostic): (int4 AS1*, int4 AS3*,
// imm offset, imm cpol).
// ---------------------------------------------------------------------------
#if defined(__HIP_DEVICE_COMPILE__) && \
    __has_builtin(__builtin_amdgcn_global_load_async_to_lds_b128)
#define USE_ASYNC_LDS 1
#else
#define USE_ASYNC_LDS 0
#endif

#define AS1P __attribute__((address_space(1))) v4i*
#define AS3P __attribute__((address_space(3))) v4i*

static __device__ __forceinline__ void async_cp16(const _Float16* g,
                                                  _Float16* l) {
#if USE_ASYNC_LDS
    __builtin_amdgcn_global_load_async_to_lds_b128((AS1P)(g), (AS3P)(l), 0, 0);
#else
    *(v8h*)l = *(const v8h*)g;
#endif
}

template <int N>
static __device__ __forceinline__ void wait_async() {
#if USE_ASYNC_LDS
#if __has_builtin(__builtin_amdgcn_s_wait_asynccnt)
    __builtin_amdgcn_s_wait_asynccnt(N);
#else
    asm volatile("s_wait_asynccnt %0" ::"i"(N) : "memory");
#endif
#endif
}

static __device__ __forceinline__ v16h mk16(v8h lo, v8h hi) {
    v16h r;
#pragma unroll
    for (int i = 0; i < 8; ++i) { r[i] = lo[i]; r[i + 8] = hi[i]; }
    return r;
}

static __device__ __forceinline__ v8f wmma_f16(v16h a, v16h b, v8f c) {
    // D = A(16x32 f16) * B(32x16 f16) + C(16x16 f32)
    return __builtin_amdgcn_wmma_f32_16x16x32_f16(false, a, false, b,
                                                  (short)0, c, false, false);
}

// ---------------------------------------------------------------------------
// Stage 1a: f32 -> f16 elementwise convert (x, y)
// ---------------------------------------------------------------------------
__global__ void cvt_f32_to_f16(const float* __restrict__ in,
                               _Float16* __restrict__ out, int n) {
    int idx = blockIdx.x * blockDim.x + threadIdx.x;  // handles 8 elements
    int i = idx * 8;
    if (i + 7 < n) {
        const float4* in4 = (const float4*)(in + i);
        float4 a = in4[0], b = in4[1];
        v8h o;
        o[0] = (_Float16)a.x; o[1] = (_Float16)a.y;
        o[2] = (_Float16)a.z; o[3] = (_Float16)a.w;
        o[4] = (_Float16)b.x; o[5] = (_Float16)b.y;
        o[6] = (_Float16)b.z; o[7] = (_Float16)b.w;
        *(v8h*)(out + i) = o;
    }
}

// ---------------------------------------------------------------------------
// Stage 1b: transpose + convert  W[R][C] f32  ->  Wt[C][R] f16
// ---------------------------------------------------------------------------
__global__ void transpose_cvt(const float* __restrict__ in,
                              _Float16* __restrict__ out, int R, int C) {
    __shared__ float tile[32][33];
    int c0 = blockIdx.x * 32;
    int r0 = blockIdx.y * 32;
    int tx = threadIdx.x & 31;
    int ty = threadIdx.x >> 5;  // 0..7
#pragma unroll
    for (int i = 0; i < 32; i += 8)
        tile[ty + i][tx] = in[(size_t)(r0 + ty + i) * C + (c0 + tx)];
    __syncthreads();
#pragma unroll
    for (int i = 0; i < 32; i += 8)
        out[(size_t)(c0 + ty + i) * R + (r0 + tx)] = (_Float16)tile[tx][ty + i];
}

// ---------------------------------------------------------------------------
// Stage 2: NT-GEMM   C[m][n] = sum_k A[m][k]*B[n][k]  (+bias), stored as
//          Ct[n][m] in f16 (WMMA accumulator columns store contiguously
//          into the transposed layout).
// 256 threads = 8 waves; tile 128(M) x 128(N); K-step 32; double-buffered
// LDS fed by async global->LDS DMA (overlaps next tile with WMMA work).
// ---------------------------------------------------------------------------
__global__ void __launch_bounds__(256)
gemm_nt_f16(const _Float16* __restrict__ A,   // [M][K]
            const _Float16* __restrict__ Bm,  // [N][K]
            _Float16* __restrict__ Ct,        // [N][M]
            const float* __restrict__ bias,   // len M (bias_on_m) or N
            int M, int N, int K, int bias_on_m) {
    constexpr int LDT = 40;  // padded LDS row stride in f16 (80B)
    __shared__ __align__(16) _Float16 As[2][128 * LDT];
    __shared__ __align__(16) _Float16 Bs[2][128 * LDT];

    const int m0 = blockIdx.x * 128;
    const int n0 = blockIdx.y * 128;
    const int tid  = threadIdx.x;
    const int lane = tid & 31;
    const int wave = tid >> 5;
    const int l16  = lane & 15;
    const int half = lane >> 4;      // 0 / 1
    const int wm = (wave & 3) * 32;  // wave's M offset inside tile
    const int wn = (wave >> 2) * 64; // wave's N offset inside tile

    v8f acc[2][4];
#pragma unroll
    for (int i = 0; i < 2; ++i)
#pragma unroll
        for (int j = 0; j < 4; ++j) acc[i][j] = v8f{};

    // global->LDS mapping: tile row = tid>>1, 32B chunk = (tid&1)*16 halfs
    const int grow = tid >> 1;
    const int goff = (tid & 1) * 16;
    const _Float16* ga = A  + (size_t)(m0 + grow) * K + goff;
    const _Float16* gb = Bm + (size_t)(n0 + grow) * K + goff;
    const int lidx = grow * LDT + goff;

    auto issue_tile = [&](int buf, int k0) {
        async_cp16(ga + k0,     &As[buf][lidx]);
        async_cp16(ga + k0 + 8, &As[buf][lidx + 8]);
        async_cp16(gb + k0,     &Bs[buf][lidx]);
        async_cp16(gb + k0 + 8, &Bs[buf][lidx + 8]);
    };

    const int NK = K / 32;
    issue_tile(0, 0);

    for (int it = 0; it < NK; ++it) {
        const int buf = it & 1;
        if (it + 1 < NK) {
            issue_tile(buf ^ 1, (it + 1) * 32);  // prefetch next tile
            wait_async<4>();  // only current tile's 4 loads must be done
        } else {
            wait_async<0>();
        }
        __syncthreads();

        const _Float16* Ab = As[buf];
        const _Float16* Bb = Bs[buf];
        v16h af[2], bf[4];
#pragma unroll
        for (int i = 0; i < 2; ++i) {
            // A-frag: low lanes k=0..7 & 16..23, high lanes k=8..15 & 24..31
            const _Float16* p = Ab + (wm + i * 16 + l16) * LDT + half * 8;
            af[i] = mk16(*(const v8h*)p, *(const v8h*)(p + 16));
        }
#pragma unroll
        for (int j = 0; j < 4; ++j) {
            // B-frag: low lanes k=0..7 & 16..23, high lanes k=8..15 & 24..31
            const _Float16* p = Bb + (wn + j * 16 + l16) * LDT + half * 8;
            bf[j] = mk16(*(const v8h*)p, *(const v8h*)(p + 16));
        }
#pragma unroll
        for (int i = 0; i < 2; ++i)
#pragma unroll
            for (int j = 0; j < 4; ++j)
                acc[i][j] = wmma_f16(af[i], bf[j], acc[i][j]);
        __syncthreads();  // all waves done reading buf before it is refilled
    }

    // Epilogue: lane holds fixed n, 8 consecutive m -> contiguous store in Ct
#pragma unroll
    for (int i = 0; i < 2; ++i)
#pragma unroll
        for (int j = 0; j < 4; ++j) {
            const int gm = m0 + wm + i * 16 + half * 8;
            const int gn = n0 + wn + j * 16 + l16;
            v8h o;
#pragma unroll
            for (int r = 0; r < 8; ++r) {
                float v = acc[i][j][r];
                v += bias_on_m ? bias[gm + r] : bias[gn];
                o[r] = (_Float16)v;
            }
            *(v8h*)(Ct + (size_t)gn * M + gm) = o;
        }
}

// ---------------------------------------------------------------------------
// Stage 3: flash attention.
//   Q  : [B*L][NP] f16 (rows q, d contiguous)        -> B-frag (Sᵀ gemm)
//   Kb : [B*L][NP] f16 (rows k2, d contiguous)       -> A-frag (Sᵀ gemm)
//   Vt : [NP][B*L] f16 (rows d, k2 contiguous)       -> B-frag (O gemm)
//   Sᵀ accumulator (m=k2, n=q) == A-frag layout of P for the O gemm, so
//   softmaxed scores become the next WMMA's A operand lane-locally.
// 256 threads (8 waves), q-block 128 (16 q / wave); K/Vᵀ tiles (64 x 64)
// double-buffered in LDS via async DMA overlapping softmax + WMMA.
// ---------------------------------------------------------------------------
__global__ void __launch_bounds__(256)
attn_flash(const _Float16* __restrict__ Q,
           const _Float16* __restrict__ Kb,
           const _Float16* __restrict__ Vt,
           const unsigned char* __restrict__ mask,  // [B][L], nonzero = pad
           float* __restrict__ out) {               // [B][L][NP] f32
    using namespace cfg;
    constexpr int LDT = 72;  // padded LDS row stride (144B)
    __shared__ __align__(16) _Float16 Ks[2][64 * LDT];
    __shared__ __align__(16) _Float16 Vs[2][64 * LDT];

    const int blk = blockIdx.x;          // b*H*8 + h*8 + qb
    const int qb  = blk & 7;
    const int h   = (blk >> 3) % H;
    const int b   = blk / (H * 8);
    const int tid  = threadIdx.x;
    const int lane = tid & 31;
    const int wave = tid >> 5;
    const int l16  = lane & 15;
    const int half = lane >> 4;
    const int q0   = qb * 128 + wave * 16;   // this wave's 16 q rows

    const float NEG_INF = -__builtin_inff();
    const float scale = 0.125f;  // 1/sqrt(64)

    // Per-thread tile-load mapping: chunk c -> (row, 16B offset)
    const int c0row = tid >> 3;              // chunk 0: rows 0..31
    const int coff  = (tid & 7) * 8;
    const _Float16* gK = Kb + (size_t)(b * L + c0row) * NP + h * DH + coff;
    const _Float16* gV = Vt + (size_t)(h * DH + c0row) * (B * L) + b * L + coff;

    auto issue_tile = [&](int buf, int k2) {
        // K tile [64 k2][64 d]: rows k2+c0row and k2+c0row+32
        async_cp16(gK + (size_t)k2 * NP, &Ks[buf][c0row * LDT + coff]);
        async_cp16(gK + (size_t)(k2 + 32) * NP, &Ks[buf][(c0row + 32) * LDT + coff]);
        // Vt tile [64 d][64 k2]: rows d=c0row and c0row+32, cols k2..k2+63
        async_cp16(gV + k2, &Vs[buf][c0row * LDT + coff]);
        async_cp16(gV + (size_t)32 * (B * L) + k2, &Vs[buf][(c0row + 32) * LDT + coff]);
    };

    // Q fragments (B operand of Sᵀ gemm), kept in registers whole kernel.
    v16h qf[2];
    {
        const _Float16* qp = Q + (size_t)(b * L + q0 + l16) * NP + h * DH;
#pragma unroll
        for (int s = 0; s < 2; ++s) {
            const _Float16* p = qp + s * 32 + half * 8;
            qf[s] = mk16(*(const v8h*)p, *(const v8h*)(p + 16));
        }
    }

    float m_i = NEG_INF, l_i = 0.0f;
    v8f o[4];
#pragma unroll
    for (int nt = 0; nt < 4; ++nt) o[nt] = v8f{};

    constexpr int NT = L / 64;  // 16 k2 tiles
    issue_tile(0, 0);

    for (int it = 0; it < NT; ++it) {
        const int buf = it & 1;
        const int k2  = it * 64;
        if (it + 1 < NT) {
            issue_tile(buf ^ 1, k2 + 64);
            wait_async<4>();
        } else {
            wait_async<0>();
        }
        __syncthreads();

        const _Float16* Kt = Ks[buf];
        const _Float16* Vb = Vs[buf];

        // Sᵀ = K · Qᵀ : 4 m-tiles (k2 sub-blocks of 16) x 2 d K-steps
        v8f st[4];
#pragma unroll
        for (int mt = 0; mt < 4; ++mt) st[mt] = v8f{};
#pragma unroll
        for (int s = 0; s < 2; ++s)
#pragma unroll
            for (int mt = 0; mt < 4; ++mt) {
                const _Float16* p = Kt + (mt * 16 + l16) * LDT + s * 32 + half * 8;
                v16h af = mk16(*(const v8h*)p, *(const v8h*)(p + 16));
                st[mt] = wmma_f16(af, qf[s], st[mt]);
            }

        // scale + padding mask + online softmax (lane owns one q = l16)
        float tmax = NEG_INF;
#pragma unroll
        for (int mt = 0; mt < 4; ++mt) {
            const int kbase = b * L + k2 + mt * 16 + 8 * half;
            const uint2 mb = *(const uint2*)(mask + kbase);  // 8 mask bytes
#pragma unroll
            for (int r = 0; r < 8; ++r) {
                unsigned byte = ((r < 4) ? (mb.x >> (8 * r))
                                         : (mb.y >> (8 * (r - 4)))) & 0xFFu;
                float sv = st[mt][r] * scale;
                if (byte) sv = NEG_INF;
                st[mt][r] = sv;
                tmax = fmaxf(tmax, sv);
            }
        }
        tmax = fmaxf(tmax, __shfl_xor(tmax, 16));
        const float m_new = fmaxf(m_i, tmax);
        const float alpha = __expf(m_i - m_new);
        float rsum = 0.0f;
#pragma unroll
        for (int mt = 0; mt < 4; ++mt)
#pragma unroll
            for (int r = 0; r < 8; ++r) {
                float p = __expf(st[mt][r] - m_new);
                st[mt][r] = p;
                rsum += p;
            }
        rsum += __shfl_xor(rsum, 16);
        l_i = l_i * alpha + rsum;
        m_i = m_new;

        // Rescale O: element r holds q = r + 8*half -> broadcast alpha per q
#pragma unroll
        for (int r = 0; r < 8; ++r) {
            float ar = __shfl(alpha, r + 8 * half);
#pragma unroll
            for (int nt = 0; nt < 4; ++nt) o[nt][r] *= ar;
        }

        // O += P · V : P A-frags built lane-locally from Sᵀ accumulators
#pragma unroll
        for (int s = 0; s < 2; ++s) {   // k2 K-steps of 32
            v16h pf;
#pragma unroll
            for (int j = 0; j < 8; ++j) {
                pf[j]     = (_Float16)st[2 * s][j];
                pf[j + 8] = (_Float16)st[2 * s + 1][j];
            }
#pragma unroll
            for (int nt = 0; nt < 4; ++nt) {
                const _Float16* p = Vb + (nt * 16 + l16) * LDT + s * 32 + half * 8;
                v16h bf = mk16(*(const v8h*)p, *(const v8h*)(p + 16));
                o[nt] = wmma_f16(pf, bf, o[nt]);
            }
        }
        __syncthreads();  // all waves done reading buf before it is refilled
    }

    // Finalize: divide by l_i (broadcast per q) and store f32
    const float linv = 1.0f / l_i;
#pragma unroll
    for (int r = 0; r < 8; ++r) {
        float lr = __shfl(linv, r + 8 * half);
        const int q = q0 + r + 8 * half;
#pragma unroll
        for (int nt = 0; nt < 4; ++nt) {
            const int d = nt * 16 + l16;
            out[(size_t)(b * L + q) * NP + h * DH + d] = o[nt][r] * lr;
        }
    }
}

// ---------------------------------------------------------------------------
// Launch
// ---------------------------------------------------------------------------
extern "C" void kernel_launch(void* const* d_in, const int* in_sizes, int n_in,
                              void* d_out, int out_size, void* d_ws, size_t ws_size,
                              hipStream_t stream) {
    using namespace cfg;
    const float* x  = (const float*)d_in[0];
    const float* y  = (const float*)d_in[1];
    const unsigned char* mask = (const unsigned char*)d_in[2];
    const float* Wq = (const float*)d_in[3];
    const float* bq = (const float*)d_in[4];
    const float* Wk = (const float*)d_in[5];
    const float* bk = (const float*)d_in[6];
    const float* Wv = (const float*)d_in[7];
    const float* bv = (const float*)d_in[8];
    float* out = (float*)d_out;

    // Workspace partition (all f16): ~67 MB total
    char* base = (char*)d_ws;
    size_t off = 0;
    auto take = [&](size_t nelem) -> _Float16* {
        _Float16* p = (_Float16*)(base + off);
        off += (nelem * sizeof(_Float16) + 255) & ~(size_t)255;
        return p;
    };
    _Float16* xh  = take((size_t)B * L * DIN);
    _Float16* yh  = take((size_t)B * L * DIN);
    _Float16* WqT = take((size_t)DIN * NP);
    _Float16* WkT = take((size_t)DIN * NP);
    _Float16* WvT = take((size_t)DIN * NP);
    _Float16* Qb  = take((size_t)B * L * NP);   // [B*L][NP]  (q rows)
    _Float16* Kbf = take((size_t)B * L * NP);   // [B*L][NP]  (k2 rows)
    _Float16* Vt  = take((size_t)NP * B * L);   // [NP][B*L]  (d rows)
    (void)ws_size; (void)in_sizes; (void)n_in; (void)out_size;

    const int nconv = B * L * DIN;  // 6291456, divisible by 2048
    cvt_f32_to_f16<<<nconv / (256 * 8), 256, 0, stream>>>(x, xh, nconv);
    cvt_f32_to_f16<<<nconv / (256 * 8), 256, 0, stream>>>(y, yh, nconv);

    dim3 tg(NP / 32, DIN / 32);
    transpose_cvt<<<tg, 256, 0, stream>>>(Wq, WqT, DIN, NP);
    transpose_cvt<<<tg, 256, 0, stream>>>(Wk, WkT, DIN, NP);
    transpose_cvt<<<tg, 256, 0, stream>>>(Wv, WvT, DIN, NP);

    // Q[q][d] = x·Wq + bq :  A=WqT (M=768=d), B=xh (N=8192=q) -> Ct = Qb
    dim3 gqk(NP / 128, (B * L) / 128);
    gemm_nt_f16<<<gqk, 256, 0, stream>>>(WqT, xh, Qb, bq, NP, B * L, DIN, 1);
    gemm_nt_f16<<<gqk, 256, 0, stream>>>(WkT, yh, Kbf, bk, NP, B * L, DIN, 1);
    // Vᵀ[d][k2] :  A=yh (M=8192=k2), B=WvT (N=768=d) -> Ct = Vt
    dim3 gv((B * L) / 128, NP / 128);
    gemm_nt_f16<<<gv, 256, 0, stream>>>(yh, WvT, Vt, bv, B * L, NP, DIN, 0);

    attn_flash<<<B * H * (L / 128), 256, 0, stream>>>(Qb, Kbf, Vt, mask, out);
}
